// RAGSequentialRec_4930622455946
// MI455X (gfx1250) — matrix operand order repaired and checked by reading, not compiled, not run on hardware
//
#include <hip/hip_runtime.h>
#include <hip/hip_bf16.h>

// ---------------------------------------------------------------------------
// Types for CDNA5 WMMA (wave32): A/B = 16 bf16 (8 VGPRs), C/D = 8 f32.
// ---------------------------------------------------------------------------
typedef __attribute__((ext_vector_type(16))) __bf16 v16bf;
typedef __attribute__((ext_vector_type(8)))  float  v8f;

#define TOPK_CAP 50

// ---------------------------------------------------------------------------
// CDNA5 async global->LDS copy (ASYNCcnt-tracked DMA, no VGPR round trip).
// lds_addr = low 32 bits of a generic pointer to __shared__ (LDS offset).
// ---------------------------------------------------------------------------
__device__ __forceinline__ void async_b128(unsigned lds_addr, const float* g) {
    asm volatile("global_load_async_to_lds_b128 %0, %1, off"
                 :: "v"(lds_addr), "v"(g)
                 : "memory");
}

// Build a 16-element bf16 fragment from two 8-float LDS chunks (RNE cvt,
// lowers to v_cvt_pk_bf16_f32; co-executes with XDL WMMAs).
__device__ __forceinline__ v16bf cvt16(const float* p_lo, const float* p_hi) {
    v16bf r;
#pragma unroll
    for (int j = 0; j < 8; ++j) r[j]     = (__bf16)p_lo[j];
#pragma unroll
    for (int j = 0; j < 8; ++j) r[j + 8] = (__bf16)p_hi[j];
    return r;
}

// ---------------------------------------------------------------------------
// Kernel 1: masked mean pooling over the sequence -> user_rep [B,D]
// ---------------------------------------------------------------------------
__global__ __launch_bounds__(256)
void pool_kernel(const int* __restrict__ seq, const float* __restrict__ E,
                 float* __restrict__ user, int S, int D) {
    const int b = blockIdx.x;
    const int t = threadIdx.x;

    int cnt = 0;
    for (int s = 0; s < S; ++s) cnt += (seq[(size_t)b * S + s] != 0);
    const float inv = 1.0f / fmaxf((float)cnt, 1.0f);

    for (int d = t; d < D; d += 256) {
        float acc = 0.0f;
        for (int s = 0; s < S; ++s) {
            int id = seq[(size_t)b * S + s];
            if (id != 0) acc += E[(size_t)(id - 1) * D + d];
        }
        user[(size_t)b * D + d] = acc * inv;
    }
}

// ---------------------------------------------------------------------------
// Kernel 2/5: C[M,N] = A[M,K] * B[N,K]^T (+bias[n]).
// fp32 tiles DMA'd into double-buffered LDS via global_load_async_to_lds_b128,
// converted to bf16 at fragment-build time, V_WMMA_F32_16X16X32_BF16 compute.
// Block tile 128x128, 256 threads = 8 waves in 2(M) x 4(N); each wave owns a
// 64x32 region = 4x2 WMMA accumulators.
// ---------------------------------------------------------------------------
#define TM 128
#define TN 128
#define TK 32
#define LPF 36   // LDS row pitch in fp32 elements (32 data + 4 pad, 16B aligned)

__device__ __forceinline__ void stage_tiles(const float* __restrict__ A,
                                            const float* __restrict__ Bm,
                                            float* Afs, float* Bfs,
                                            int m0, int n0, int M, int N, int K,
                                            int k0, int tid) {
    // 8 async b128 copies per thread: 128x32 fp32 for A and for B.
    // Source rows are clamped (never stored for OOB m/n), keeping EXEC full.
#pragma unroll
    for (int it = 0; it < 4; ++it) {
        int q = tid + it * 256, r = q >> 3, c4 = (q & 7) << 2;
        int mrow = min(m0 + r, M - 1);
        async_b128((unsigned)(size_t)&Afs[r * LPF + c4],
                   A + (size_t)mrow * K + k0 + c4);
    }
#pragma unroll
    for (int it = 0; it < 4; ++it) {
        int q = tid + it * 256, r = q >> 3, c4 = (q & 7) << 2;
        int n = min(n0 + r, N - 1);
        async_b128((unsigned)(size_t)&Bfs[r * LPF + c4],
                   Bm + (size_t)n * K + k0 + c4);
    }
}

__global__ __launch_bounds__(256)
void gemm_bf16(const float* __restrict__ A, const float* __restrict__ Bm,
               const float* __restrict__ bias, float* __restrict__ C,
               int M, int N, int K) {
    __shared__ float Afs[2][TM * LPF];   // 2 x 18 KB
    __shared__ float Bfs[2][TN * LPF];   // 2 x 18 KB

    const int m0  = blockIdx.y * TM;
    const int n0  = blockIdx.x * TN;
    const int tid = threadIdx.x;
    const int wid  = tid >> 5;
    const int lane = tid & 31;
    const int hl = lane >> 4;     // lane half (0/1)
    const int nl = lane & 15;     // M-row / N-col within 16x16 tile
    const int wm = wid >> 2;      // 0..1
    const int wn = wid & 3;       // 0..3

    v8f acc[4][2];
#pragma unroll
    for (int i = 0; i < 4; ++i)
#pragma unroll
        for (int j = 0; j < 2; ++j) acc[i][j] = (v8f){0,0,0,0,0,0,0,0};

    // prologue: stage k-tile 0 into buffer 0
    stage_tiles(A, Bm, &Afs[0][0], &Bfs[0][0], m0, n0, M, N, K, 0, tid);

    int sb = 0;
    for (int k0 = 0; k0 < K; k0 += TK, sb ^= 1) {
        if (k0 + TK < K) {
            // issue next k-tile into the other buffer, then retire current
            // batch (ASYNCcnt 'done' returns in order => <=8 means batch sb done)
            stage_tiles(A, Bm, &Afs[sb ^ 1][0], &Bfs[sb ^ 1][0],
                        m0, n0, M, N, K, k0 + TK, tid);
            asm volatile("s_wait_asynccnt 0x8" ::: "memory");
        } else {
            asm volatile("s_wait_asynccnt 0x0" ::: "memory");
        }
        __syncthreads();   // all waves' DMA for buffer sb visible

        const float* Ab = &Afs[sb][0];
        const float* Bb = &Bfs[sb][0];

        // fragments per ISA 7.12.2 16-bit layouts, with on-the-fly f32->bf16
        v16bf afr[4], bfr[2];
#pragma unroll
        for (int i = 0; i < 4; ++i) {
            const float* base = Ab + (wm * 64 + i * 16 + nl) * LPF;
            // A: K = 8*hl + j (j<8) ; K = 16 + 8*hl + (j-8) (j>=8)
            afr[i] = cvt16(base + 8 * hl, base + 16 + 8 * hl);
        }
#pragma unroll
        for (int j = 0; j < 2; ++j) {
            const float* base = Bb + (wn * 32 + j * 16 + nl) * LPF;
            // B: K = 8*hl + j (j<8) ; K = 16 + 8*hl + (j-8) (j>=8)
            bfr[j] = cvt16(base + 8 * hl, base + 16 + 8 * hl);
        }
#pragma unroll
        for (int i = 0; i < 4; ++i)
#pragma unroll
            for (int j = 0; j < 2; ++j)
                acc[i][j] = __builtin_amdgcn_wmma_f32_16x16x32_bf16(
                    false, afr[i], false, bfr[j], (short)0, acc[i][j],
                    false, false);
        __syncthreads();   // buffer sb free for reuse (DMA of iter+2)
    }

    // ---- epilogue: C/D layout M = 8*hl + vgpr, N = nl ----
#pragma unroll
    for (int j = 0; j < 2; ++j) {
        int n = n0 + wn * 32 + j * 16 + nl;
        if (n >= N) continue;
        float bb = bias ? bias[n] : 0.0f;
#pragma unroll
        for (int i = 0; i < 4; ++i) {
            int mbase = m0 + wm * 64 + i * 16 + 8 * hl;
#pragma unroll
            for (int r = 0; r < 8; ++r) {
                int mm = mbase + r;
                if (mm < M) C[(size_t)mm * N + n] = acc[i][j][r] + bb;
            }
        }
    }
}

// ---------------------------------------------------------------------------
// Kernel 3: per-row top-k.  64 threads; each keeps a private top-50 of its
// stride, candidates merged in LDS, then Kq argmax-selection rounds.
// ---------------------------------------------------------------------------
__global__ __launch_bounds__(64)
void topk_kernel(const float* __restrict__ scores, int* __restrict__ tidx,
                 const int* __restrict__ topk_ptr, int N) {
    __shared__ float cval[64 * TOPK_CAP];
    __shared__ int   cidx[64 * TOPK_CAP];
    __shared__ float rmax[64];
    __shared__ int   rpos[64];

    const int b  = blockIdx.x;
    const int t  = threadIdx.x;
    const int Kq = min(*topk_ptr, TOPK_CAP);
    const float NEG = -3.4e38f;

    float v[TOPK_CAP]; int ix[TOPK_CAP];
#pragma unroll
    for (int k = 0; k < TOPK_CAP; ++k) { v[k] = NEG; ix[k] = 0; }
    float vmin = NEG; int mpos = 0;

    const float* row = scores + (size_t)b * N;
    for (int n = t; n < N; n += 64) {
        float s = row[n];
        if (s > vmin) {
            v[mpos] = s; ix[mpos] = n;
            vmin = v[0]; mpos = 0;
            for (int k = 1; k < TOPK_CAP; ++k)
                if (v[k] < vmin) { vmin = v[k]; mpos = k; }
        }
    }
    for (int k = 0; k < TOPK_CAP; ++k) {
        cval[t * TOPK_CAP + k] = v[k];
        cidx[t * TOPK_CAP + k] = ix[k];
    }
    __syncthreads();

    for (int kk = 0; kk < Kq; ++kk) {
        float bm = NEG; int bp = 0;
        for (int q = t; q < 64 * TOPK_CAP; q += 64)
            if (cval[q] > bm) { bm = cval[q]; bp = q; }
        rmax[t] = bm; rpos[t] = bp;
        __syncthreads();
        if (t == 0) {
            float m = rmax[0]; int p = rpos[0];
            for (int i = 1; i < 64; ++i)
                if (rmax[i] > m) { m = rmax[i]; p = rpos[i]; }
            tidx[(size_t)b * TOPK_CAP + kk] = cidx[p];
            cval[p] = NEG;
        }
        __syncthreads();
    }
}

// ---------------------------------------------------------------------------
// Kernel 4: retrieved mean + gated fusion + LayerNorm -> fused [B,D]
// ---------------------------------------------------------------------------
__global__ __launch_bounds__(256)
void fuse_kernel(const int* __restrict__ tidx, const float* __restrict__ E,
                 const float* __restrict__ user, const float* __restrict__ fW,
                 const float* __restrict__ fb, const float* __restrict__ g_,
                 const float* __restrict__ b_, float* __restrict__ fused,
                 const int* __restrict__ topk_ptr, int D, int N) {
    __shared__ float fin[2048];   // [ user(D) | retrieved(D) ]
    __shared__ float fx[1024];    // pre-LN fused
    __shared__ float red[256], red2[256];

    const int b = blockIdx.x;
    const int t = threadIdx.x;
    const int wid = t >> 5, lane = t & 31;
    const int Kq = min(*topk_ptr, TOPK_CAP);
    const float invK = 1.0f / (float)Kq;

    for (int d = t; d < D; d += 256) {
        float acc = 0.0f;
        for (int k = 0; k < Kq; ++k) {
            int id = tidx[(size_t)b * TOPK_CAP + k];
            acc += E[(size_t)id * D + d];
        }
        fin[D + d] = acc * invK;                 // retrieved
        fin[d]     = user[(size_t)b * D + d];    // user_rep
    }
    __syncthreads();

    // wave-cooperative gate GEMV: wave `wid` handles rows d = wid, wid+8, ...
    const int D2 = 2 * D;
    for (int d = wid; d < D; d += 8) {
        const float* w = fW + (size_t)d * D2;
        float p = 0.0f;
        for (int j = lane; j < D2; j += 32) p += fin[j] * w[j];
#pragma unroll
        for (int off = 16; off > 0; off >>= 1) p += __shfl_xor(p, off, 32);
        if (lane == 0) {
            float g = 1.0f / (1.0f + __expf(-(p + fb[d])));
            fx[d] = g * fin[d] + (1.0f - g) * fin[D + d];
        }
    }
    __syncthreads();

    // LayerNorm over D
    float s = 0.0f, s2 = 0.0f;
    for (int d = t; d < D; d += 256) { float x = fx[d]; s += x; s2 += x * x; }
    red[t] = s; red2[t] = s2;
    __syncthreads();
    for (int off = 128; off > 0; off >>= 1) {
        if (t < off) { red[t] += red[t + off]; red2[t] += red2[t + off]; }
        __syncthreads();
    }
    const float mu  = red[0] / (float)D;
    const float var = red2[0] / (float)D - mu * mu;
    const float rs  = __frsqrt_rn(var + 1e-5f);

    for (int d = t; d < D; d += 256)
        fused[(size_t)b * D + d] = (fx[d] - mu) * rs * g_[d] + b_[d];
}

// ---------------------------------------------------------------------------
// Launch
// ---------------------------------------------------------------------------
extern "C" void kernel_launch(void* const* d_in, const int* in_sizes, int n_in,
                              void* d_out, int out_size, void* d_ws, size_t ws_size,
                              hipStream_t stream) {
    const int*   seq   = (const int*)  d_in[0];
    const float* E     = (const float*)d_in[1];
    const float* fW    = (const float*)d_in[2];
    const float* fb    = (const float*)d_in[3];
    const float* lng   = (const float*)d_in[4];
    const float* lnb   = (const float*)d_in[5];
    const float* pW    = (const float*)d_in[6];
    const float* pb    = (const float*)d_in[7];
    const int*   topk  = (const int*)  d_in[8];

    const int D = in_sizes[3];          // 512
    const int N = in_sizes[7];          // 100000
    const int B = out_size / N;         // 1024
    const int S = in_sizes[0] / B;      // 200

    float* user   = (float*)d_ws;                        // B*D
    float* fused  = user + (size_t)B * D;                // B*D
    int*   tidx   = (int*)(fused + (size_t)B * D);       // B*TOPK_CAP
    float* scores = (float*)d_out;                       // staged, overwritten by logits

    dim3 ggrid((N + TN - 1) / TN, (B + TM - 1) / TM);

    pool_kernel<<<B, 256, 0, stream>>>(seq, E, user, S, D);
    gemm_bf16 <<<ggrid, 256, 0, stream>>>(user, E, nullptr, scores, B, N, D);
    topk_kernel<<<B, 64, 0, stream>>>(scores, tidx, topk, N);
    fuse_kernel<<<B, 256, 0, stream>>>(tidx, E, user, fW, fb, lng, lnb, fused, topk, D, N);
    gemm_bf16 <<<ggrid, 256, 0, stream>>>(fused, pW, pb, (float*)d_out, B, N, D);
}
